// GNN_10453950399131
// MI455X (gfx1250) — matrix operand order repaired and checked by reading, not compiled, not run on hardware
//
#include <hip/hip_runtime.h>
#include <hip/hip_bf16.h>

// ---------------------------------------------------------------------------
// GAT (3-layer, dense adjacency) for MI455X / gfx1250, wave32 + WMMA bf16.
// N=4096, F_IN=512, D_H=64, HEADS=8, D_OUT=64.
//
// out[i,h,:] = (sum_j e_ij * x_j) / (sum_j e_ij): denominator comes free from
// a ones-column in the transposed feature matrix (5th WMMA N-tile).
// Layout choices: x_augT[80][N] and Wt[64][512] are stored transposed so every
// WMMA B-fragment is two contiguous global_load_b128 per lane. Adjacency
// chunks are staged to LDS via async global->LDS DMA (ASYNCcnt) and shared by
// all head-waves of a block, double-buffered.
// ---------------------------------------------------------------------------

#define NN     4096
#define DH     64

typedef __bf16          v16bf __attribute__((ext_vector_type(16)));
typedef unsigned short  v16us __attribute__((ext_vector_type(16)));
typedef float           v8f   __attribute__((ext_vector_type(8)));
typedef __attribute__((address_space(3))) float lds_f32;

__device__ __forceinline__ unsigned short f32_to_bf16(float f) {
    union { float f; unsigned int u; } c; c.f = f;
    unsigned int u = c.u;
    unsigned int r = u + 0x7FFFu + ((u >> 16) & 1u);   // round-to-nearest-even
    return (unsigned short)(r >> 16);
}

union B16x16 { uint4 q[2]; v16us v; };

// ------------------------------- casts -------------------------------------
__global__ void cast_f32_bf16(const float* __restrict__ in,
                              unsigned short* __restrict__ out, int n) {
    int i = blockIdx.x * blockDim.x + threadIdx.x;
    if (i < n) out[i] = f32_to_bf16(in[i]);
}

// W[512,64] f32 -> Wt[64,512] bf16 (transposed for contiguous B-fragments)
__global__ void cast_transpose_w(const float* __restrict__ W,
                                 unsigned short* __restrict__ Wt) {
    int i = blockIdx.x * blockDim.x + threadIdx.x;   // i = k*64 + n
    if (i >= 512 * 64) return;
    int k = i >> 6, n = i & 63;
    Wt[(size_t)n * 512 + k] = f32_to_bf16(W[i]);
}

// ------------------------- GEMM: C[M,64] = A[M,K] @ W[K,64] ----------------
// A row-major bf16, Wt = W^T [64,K] bf16, f32 out. One wave per 16x16 tile.
__global__ void wmma_gemm_bf16(const unsigned short* __restrict__ A,
                               const unsigned short* __restrict__ Wt,
                               float* __restrict__ C, int K) {
    const int lane = threadIdx.x & 31;
    const int wave = threadIdx.x >> 5;        // column tile 0..3
    const int m0   = blockIdx.x * 16;
    const int n0   = wave * 16;
    const int m    = lane & 15;
    const int half = lane >> 4;
    const int n    = m;

    v8f acc = {0.f,0.f,0.f,0.f,0.f,0.f,0.f,0.f};

    for (int k0 = 0; k0 < K; k0 += 32) {
        // A 16x32 bf16 fragment: lane holds K runs {half*8..+7} and {16+half*8..+7}
        const unsigned short* ap = A + (size_t)(m0 + m) * K + k0 + half * 8;
        B16x16 ua;
        ua.q[0] = *(const uint4*)(ap);
        ua.q[1] = *(const uint4*)(ap + 16);
        // B 32x16 fragment: lane = column n, K runs {half*8..+7} and {16+half*8..+7} in Wt
        const unsigned short* bp = Wt + (size_t)(n0 + n) * 512 + k0 + half * 8;
        B16x16 ub;
        ub.q[0] = *(const uint4*)(bp);
        ub.q[1] = *(const uint4*)(bp + 16);
        acc = __builtin_amdgcn_wmma_f32_16x16x32_bf16(
                false, __builtin_bit_cast(v16bf, ua.v),
                false, __builtin_bit_cast(v16bf, ub.v),
                (short)0, acc, false, false);
    }
#pragma unroll
    for (int v = 0; v < 8; ++v)               // C/D: row = v + 8*half, col = n
        C[(size_t)(m0 + v + 8 * half) * 64 + n0 + n] = acc[v];
}

// ----------------- per-node attn logits + transposed bf16 features ---------
// x_augT[80][N]: rows 0..63 = x^T, row 64 = 1.0 (denominator), 65..79 = 0.
// elT/erT [H,N] so aggregation reads er contiguously along j.
__global__ void gat_prep(const float* __restrict__ x,          // [N,64]
                         const float* __restrict__ al,         // [64,H]
                         const float* __restrict__ ar,         // [64,H]
                         unsigned short* __restrict__ x_augT,  // [80,N]
                         float* __restrict__ elT,              // [H,N]
                         float* __restrict__ erT,              // [H,N]
                         int H) {
    int i = blockIdx.x * blockDim.x + threadIdx.x;
    if (i >= NN) return;
    float xv[DH];
#pragma unroll
    for (int d = 0; d < DH; ++d) xv[d] = x[(size_t)i * DH + d];
    for (int h = 0; h < H; ++h) {
        float el = 0.f, er = 0.f;
#pragma unroll
        for (int d = 0; d < DH; ++d) {
            el += xv[d] * al[d * H + h];
            er += xv[d] * ar[d * H + h];
        }
        elT[(size_t)h * NN + i] = el;
        erT[(size_t)h * NN + i] = er;
    }
#pragma unroll
    for (int d = 0; d < DH; ++d) x_augT[(size_t)d * NN + i] = f32_to_bf16(xv[d]);
    x_augT[(size_t)64 * NN + i] = 0x3F80;      // 1.0 bf16
#pragma unroll
    for (int d = 65; d < 80; ++d) x_augT[(size_t)d * NN + i] = 0;
}

// -------------------------- fused masked-softmax aggregate -----------------
// Block = WAVES waves; wave h owns head h of one 16-row i-tile. The 16x32 f32
// adjacency chunk is DMA'd to LDS (async global->LDS, double buffered) once
// per block and shared by all heads. Per chunk: build bf16 A-fragment of
// e_ij = mask*exp(lrelu(el_i+er_j)), then 5 WMMAs accumulate num[16x64]+denom.
template <int WAVES>
__global__ void gat_aggregate(const float* __restrict__ adj,      // [N,N]
                              const unsigned short* __restrict__ x_augT, // [80,N]
                              const float* __restrict__ elT,      // [H,N]
                              const float* __restrict__ erT,      // [H,N]
                              const float* __restrict__ bias,     // [64]
                              float* __restrict__ outF,           // [N,H*64] or null
                              unsigned short* __restrict__ outB,  // bf16 or null
                              int act) {
    constexpr int H      = WAVES;
    constexpr int CHUNKS = NN / 32;
    constexpr int EPT    = 512 / (WAVES * 32);   // floats staged per thread
    constexpr int OPS    = EPT / 2;              // async b64 ops per thread/chunk

    __shared__ float adjT[2][16][32];

    const int lane = threadIdx.x & 31;
    const int h    = threadIdx.x >> 5;
    const int i0   = blockIdx.x * 16;
    const int m    = lane & 15;
    const int half = lane >> 4;
    const int n    = m;

    const float  el_i = elT[(size_t)h * NN + (i0 + m)];
    const float* erH  = erT + (size_t)h * NN;

    v8f acc[5];
#pragma unroll
    for (int t = 0; t < 5; ++t) acc[t] = (v8f){0.f,0.f,0.f,0.f,0.f,0.f,0.f,0.f};

    // async-DMA one 16x32 f32 adj chunk into LDS buffer `buf`
    auto issue = [&](int c, int buf) {
#pragma unroll
        for (int q = 0; q < OPS; ++q) {
            int f = threadIdx.x * EPT + 2 * q;         // flat float index
            int r = f >> 5, cc = f & 31;
            const float* g = adj + (size_t)(i0 + r) * NN + (size_t)c * 32 + cc;
            unsigned loff = (unsigned)(unsigned long long)(lds_f32*)(&adjT[buf][r][cc]);
            asm volatile("global_load_async_to_lds_b64 %0, %1, off"
                         :: "v"(loff), "v"(g) : "memory");
        }
    };

    issue(0, 0);
    for (int c = 0; c < CHUNKS; ++c) {
        const int j0 = c * 32;
        const int buf = c & 1;
        if (c + 1 < CHUNKS) {
            issue(c + 1, buf ^ 1);
            asm volatile("s_wait_asynccnt %0" :: "n"(OPS) : "memory");
        } else {
            asm volatile("s_wait_asynccnt 0" ::: "memory");
        }
        __syncthreads();                       // all waves' DMA visible

        // A-fragment: lane m, K runs {half*8..+7} and {16+half*8..+7}
        float mv[16], ev[16];
        {
            const float4* a0 = (const float4*)&adjT[buf][m][half * 8];
            const float4* a1 = (const float4*)&adjT[buf][m][16 + half * 8];
            *(float4*)&mv[0]  = a0[0]; *(float4*)&mv[4]  = a0[1];
            *(float4*)&mv[8]  = a1[0]; *(float4*)&mv[12] = a1[1];
            const float4* e0 = (const float4*)(erH + j0 + half * 8);
            const float4* e1 = (const float4*)(erH + j0 + 16 + half * 8);
            *(float4*)&ev[0]  = e0[0]; *(float4*)&ev[4]  = e0[1];
            *(float4*)&ev[8]  = e1[0]; *(float4*)&ev[12] = e1[1];
        }
        v16us a_us;
#pragma unroll
        for (int q = 0; q < 16; ++q) {
            float s = el_i + ev[q];
            s = (s > 0.f) ? s : 0.2f * s;                  // leaky_relu(0.2)
            float w = (mv[q] != 0.f) ? __expf(s) : 0.f;    // mask * exp
            a_us[q] = f32_to_bf16(w);
        }
        v16bf a = __builtin_bit_cast(v16bf, a_us);

#pragma unroll
        for (int t = 0; t < 5; ++t) {
            // B-fragment: lane column n, K runs {half*8..+7} and {16+half*8..+7} in x_augT
            const unsigned short* bp =
                x_augT + (size_t)(t * 16 + n) * NN + j0 + half * 8;
            B16x16 ub;
            ub.q[0] = *(const uint4*)(bp);
            ub.q[1] = *(const uint4*)(bp + 16);
            acc[t] = __builtin_amdgcn_wmma_f32_16x16x32_bf16(
                        false, a, false, __builtin_bit_cast(v16bf, ub.v),
                        (short)0, acc[t], false, false);
        }
        __syncthreads();                       // buffer reuse barrier
    }

    // acc[4] col 0 (lanes 0/16) = denom per row; broadcast across the row.
#pragma unroll
    for (int v = 0; v < 8; ++v) {
        float den = __shfl(acc[4][v], lane & 16, 32);
        den = fmaxf(den, 1e-12f);
        int row = i0 + v + 8 * half;
#pragma unroll
        for (int t = 0; t < 4; ++t) {
            float o = acc[t][v] / den + bias[t * 16 + n];
            if (act) o = (o > 0.f) ? o : (__expf(o) - 1.f);   // ELU
            size_t idx = (size_t)row * ((size_t)H * 64) + (size_t)h * 64 + t * 16 + n;
            if (outF) outF[idx] = o;
            else      outB[idx] = f32_to_bf16(o);
        }
    }
}

// ---------------------------------------------------------------------------
extern "C" void kernel_launch(void* const* d_in, const int* in_sizes, int n_in,
                              void* d_out, int out_size, void* d_ws, size_t ws_size,
                              hipStream_t stream) {
    const float* adj  = (const float*)d_in[0];
    const float* feat = (const float*)d_in[1];
    const float* W[3]  = { (const float*)d_in[2], (const float*)d_in[6],  (const float*)d_in[10] };
    const float* al[3] = { (const float*)d_in[3], (const float*)d_in[7],  (const float*)d_in[11] };
    const float* ar[3] = { (const float*)d_in[4], (const float*)d_in[8],  (const float*)d_in[12] };
    const float* bs[3] = { (const float*)d_in[5], (const float*)d_in[9],  (const float*)d_in[13] };

    // workspace layout (all 256B aligned)
    char* ws = (char*)d_ws;
    unsigned short* hb    = (unsigned short*)(ws);                      // [4096,512] bf16 (4 MB)
    unsigned short* Wt    = (unsigned short*)(ws + (4u<<20));           // [64,512]  bf16 (64 KB)
    float*          x     = (float*)         (ws + (4u<<20) + (1u<<16));// [4096,64] f32  (1 MB)
    unsigned short* xaugT = (unsigned short*)(ws + (5u<<20) + (1u<<16));// [80,4096] bf16 (640 KB)
    float*          elT   = (float*)         (ws + (6u<<20));           // [8,4096]  f32  (128 KB)
    float*          erT   = (float*)         (ws + (6u<<20) + (1u<<17));// [8,4096]  f32  (128 KB)

    // features -> bf16 once (row-major A for the GEMM)
    cast_f32_bf16<<<(NN*512 + 255)/256, 256, 0, stream>>>(feat, hb, NN*512);

    for (int L = 0; L < 3; ++L) {
        const int H = (L == 2) ? 1 : 8;
        cast_transpose_w<<<(512*64 + 255)/256, 256, 0, stream>>>(W[L], Wt);
        wmma_gemm_bf16<<<NN/16, 128, 0, stream>>>(hb, Wt, x, 512);
        gat_prep<<<NN/256, 256, 0, stream>>>(x, al[L], ar[L], xaugT, elT, erT, H);
        if (L < 2) {
            // ELU + write bf16 directly as next layer's GEMM input
            gat_aggregate<8><<<NN/16, 256, 0, stream>>>(adj, xaugT, elT, erT, bs[L],
                                                        nullptr, hb, 1);
        } else {
            gat_aggregate<1><<<NN/16, 32, 0, stream>>>(adj, xaugT, elT, erT, bs[L],
                                                       (float*)d_out, nullptr, 0);
        }
    }
}
